// KPConvLayer_23450521436528
// MI455X (gfx1250) — matrix operand order repaired and checked by reading, compiled, not run on hardware
//
#include <hip/hip_runtime.h>
#include <hip/hip_bf16.h>

// ---- CDNA5 vector types --------------------------------------------------
typedef __attribute__((ext_vector_type(16))) __bf16 v16bf;
typedef __attribute__((ext_vector_type(8)))  __bf16 v8bf;
typedef __attribute__((ext_vector_type(2)))  __bf16 v2bf;
typedef __attribute__((ext_vector_type(8)))  float  v8f;
typedef __attribute__((ext_vector_type(4)))  float  v4f;
typedef __attribute__((ext_vector_type(2)))  float  v2f;
typedef __attribute__((ext_vector_type(4)))  unsigned int v4u;
typedef __attribute__((ext_vector_type(8)))  int    v8i;
typedef __attribute__((ext_vector_type(4)))  int    v4i;

#define B_    4
#define M_    16384
#define KNB   32
#define DIN   64
#define DOUT  64
#define KP    15
#define KFLAT (KP * DIN)        // 960 = 30 * 32  (exact WMMA K chunks)
#define NKK   (KFLAT / 32)      // 30
#define FKSTRIDE 968            // 960 + 8 halves pad: 16B-aligned, conflict-free

// ---------------------------------------------------------------------------
// Prelude: pack W [15][64][64] f32 -> bf16 in WMMA B-operand lane layout.
// WB element t = ((kk*4 + nt)*32 + L)*16 + j holds W[p][d][e] with
//   K = kk*32 + (L<16 ? j : 16+j),  p = K/64, d = K%64,  e = nt*16 + (L&15)
// so each lane L reads its v16bf B fragment as 32 contiguous bytes.
// ---------------------------------------------------------------------------
__global__ __launch_bounds__(256)
void pack_w_kernel(const float* __restrict__ W, __bf16* __restrict__ WB) {
    const int t = blockIdx.x * 256 + threadIdx.x;
    if (t >= NKK * 4 * 32 * 16) return;
    const int j  = t & 15;
    const int L  = (t >> 4) & 31;
    const int nt = (t >> 9) & 3;
    const int kk = t >> 11;
    const int K  = kk * 32 + ((L < 16) ? j : (16 + j));
    const int p  = K >> 6;
    const int d  = K & 63;
    const int e  = nt * 16 + (L & 15);
    WB[t] = (__bf16)W[(p * DIN + d) * DOUT + e];
}

// ---------------------------------------------------------------------------
// TDM gather: fetch 16 feature rows (256B each, data_size=8B) selected by
// 16-bit row indices into LDS at ldsOff.  Gather-mode D# per cdna5 ISA §8.
// clang-23 toolchain: 6-arg builtin (g0 v4u, g1 v8i, g2 v4i, g3 v4i, v8i, cpol)
// ---------------------------------------------------------------------------
__device__ __forceinline__ void tdm_gather16(const float* fbase, const int* nrow,
                                             unsigned ldsOff) {
    unsigned long long ga = (unsigned long long)(const void*)fbase;
    v4u g0;
    g0.x = 1u | (1u << 31);                              // count=1, gather_mode=1, 16-bit idx
    g0.y = ldsOff;                                       // LDS byte address
    g0.z = (unsigned)ga;                                 // global_addr[31:0]
    g0.w = ((unsigned)(ga >> 32) & 0x1FFFFFFu) | (2u << 30);  // addr[56:32] | type=2
    v8i g1;
    g1[0] = (int)(3u << 16);         // workgroup_mask=0, data_size=3 (8B)
    g1[1] = (int)(32u << 16);        // tensor_dim0 = 32 elements (256B row)
    g1[2] = (int)(16384u << 16);     // tensor_dim0 hi=0 | tensor_dim1 lo = M
    g1[3] = (int)(32u << 16);        // tensor_dim1 hi=0 | tile_dim0 = 32
    g1[4] = 16;                      // tile_dim1 = #valid indices = 16
    g1[5] = 32;                      // tensor_dim0_stride = 32 elements
    g1[6] = 0;
    g1[7] = 0;
    v4i g2, g3;
#pragma unroll
    for (int i = 0; i < 4; ++i) {
        g2[i] = (int)(((unsigned)nrow[2 * i] & 0xFFFFu) |
                      ((unsigned)nrow[2 * i + 1] << 16));
        g3[i] = (int)(((unsigned)nrow[8 + 2 * i] & 0xFFFFu) |
                      ((unsigned)nrow[8 + 2 * i + 1] << 16));
    }
    const v8i gpad = (v8i){0, 0, 0, 0, 0, 0, 0, 0};
    __builtin_amdgcn_tensor_load_to_lds(g0, g1, g2, g3, gpad, 0);
}

// ---------------------------------------------------------------------------
// Main kernel: one wave (32 lanes) per tile of 16 center points.
//  Stage 1: per-lane influence weights h[k][p] -> LDS  (lane = neighbor k)
//  Stage 2: Fk[m][p][d] += h * Fn  (VALU f32, Fn rows staged by TDM gather)
//  Stage 3: Out[16,64] = FkFlat[16,960] @ WB[960,64]  via v_wmma bf16
// ---------------------------------------------------------------------------
__global__ __launch_bounds__(32)
void kpconv_kernel(const float* __restrict__ X, const float* __restrict__ F,
                   const int* __restrict__ Nn, const float* __restrict__ Q,
                   const v8bf* __restrict__ WB, float* __restrict__ Out) {
    const int lane = threadIdx.x;
    const int b    = blockIdx.y;
    const int m0   = blockIdx.x * 16;

    __shared__ __align__(16) __bf16 sFk[16 * FKSTRIDE];   // 30976 B
    __shared__ __align__(16) float  sFn[2][KNB * DIN];    // 16384 B (double buf)
    __shared__ __align__(16) float  sH[KNB * 16];         //  2048 B

    const float* fbase = F + (unsigned long long)b * M_ * DIN;

    // prime buffer 0 with point m0's 32 neighbor rows (2 TDM gathers)
    {
        const int* nrow = Nn + ((unsigned long long)b * M_ + m0) * KNB;
        unsigned base = (unsigned)(unsigned long long)&sFn[0][0];
        tdm_gather16(fbase, nrow, base);
        tdm_gather16(fbase, nrow + 16, base + 16 * DIN * 4);
    }

    for (int m = 0; m < 16; ++m) {
        const int mi  = m0 + m;
        const int buf = m & 1;
        const unsigned long long rowb = (unsigned long long)b * M_ + mi;

        // ---- Stage 1: lane = neighbor k, compute 15 influence weights ----
        const int idx = Nn[rowb * KNB + lane];
        const float cx = X[rowb * 3 + 0];
        const float cy = X[rowb * 3 + 1];
        const float cz = X[rowb * 3 + 2];
        const float* xr = X + ((unsigned long long)b * M_ + idx) * 3;
        const float rx = xr[0] - cx, ry = xr[1] - cy, rz = xr[2] - cz;

        v4f h4[4];
#pragma unroll
        for (int p = 0; p < 16; ++p) {
            float h = 0.0f;
            if (p < KP) {
                const float dx = rx - Q[p * 3 + 0];
                const float dy = ry - Q[p * 3 + 1];
                const float dz = rz - Q[p * 3 + 2];
                const float dd = __builtin_sqrtf(dx * dx + dy * dy + dz * dz);
                h = fmaxf(0.0f, 1.0f - dd);
            }
            h4[p >> 2][p & 3] = h;
        }
#pragma unroll
        for (int q4 = 0; q4 < 4; ++q4)
            *(v4f*)&sH[lane * 16 + q4 * 4] = h4[q4];

        // ---- prefetch next point's neighbor rows into the other buffer ----
        if (m + 1 < 16) {
            const int* nrow = Nn + (rowb + 1) * KNB;
            unsigned base = (unsigned)(unsigned long long)&sFn[buf ^ 1][0];
            tdm_gather16(fbase, nrow, base);
            tdm_gather16(fbase, nrow + 16, base + 16 * DIN * 4);
            __builtin_amdgcn_s_wait_tensorcnt(2);  // this buffer's 2 done
        } else {
            __builtin_amdgcn_s_wait_tensorcnt(0);
        }
        __syncthreads();  // order sH stores vs cross-lane sH loads

        // ---- Stage 2: Fk accumulation. lane owns d = {2*lane, 2*lane+1} ----
        v2f acc[KP];
#pragma unroll
        for (int p = 0; p < KP; ++p) acc[p] = (v2f){0.0f, 0.0f};

        for (int k = 0; k < KNB; ++k) {
            const v2f f2 = *(const v2f*)&sFn[buf][k * DIN + 2 * lane];
            const v4f hh0 = *(const v4f*)&sH[k * 16 + 0];
            const v4f hh1 = *(const v4f*)&sH[k * 16 + 4];
            const v4f hh2 = *(const v4f*)&sH[k * 16 + 8];
            const v4f hh3 = *(const v4f*)&sH[k * 16 + 12];
            const v4f hh[4] = {hh0, hh1, hh2, hh3};
#pragma unroll
            for (int p = 0; p < KP; ++p) {
                const float h = hh[p >> 2][p & 3];
                acc[p].x += h * f2.x;
                acc[p].y += h * f2.y;
            }
        }

        // ---- write Fk row (bf16) into WMMA-A staging buffer ----
#pragma unroll
        for (int p = 0; p < KP; ++p) {
            v2bf pk;
            pk.x = (__bf16)acc[p].x;
            pk.y = (__bf16)acc[p].y;
            *(v2bf*)&sFk[m * FKSTRIDE + p * DIN + 2 * lane] = pk;
        }
    }

    __syncthreads();

    // ---- Stage 3: Out[16,64] = FkFlat[16,960] @ Wflat[960,64] via WMMA ----
    const int row = lane & 15;
    const int hi  = lane >> 4;
    v8f c[4];
#pragma unroll
    for (int nt = 0; nt < 4; ++nt) c[nt] = (v8f){};

    const __bf16* arow = sFk + row * FKSTRIDE;
    for (int kk = 0; kk < NKK; ++kk) {
        // A fragment (contiguous K, matching WB packing): lanes<16 take K {0..15},
        // lanes>=16 take K {16..31} of this 32-wide chunk.
        const __bf16* ab = arow + kk * 32 + hi * 16;
        v16bf A;
        *(v8bf*)&A       = *(const v8bf*)(ab);
        *((v8bf*)&A + 1) = *(const v8bf*)(ab + 8);
#pragma unroll
        for (int nt = 0; nt < 4; ++nt) {
            const v8bf* bp = WB + (unsigned)((kk * 4 + nt) * 32 + lane) * 2;
            v16bf Bm;
            *(v8bf*)&Bm       = bp[0];
            *((v8bf*)&Bm + 1) = bp[1];
            c[nt] = __builtin_amdgcn_wmma_f32_16x16x32_bf16(
                false, A, false, Bm, (short)0, c[nt], false, false);
        }
    }

    // ---- store: C/D layout -> point row r + hi*8, column nt*16 + row ----
    float* orow = Out + ((unsigned long long)b * M_ + m0) * DOUT;
#pragma unroll
    for (int nt = 0; nt < 4; ++nt) {
#pragma unroll
        for (int r = 0; r < 8; ++r) {
            orow[(unsigned long long)(r + hi * 8) * DOUT + nt * 16 + row] =
                c[nt][r];
        }
    }
}

// ---------------------------------------------------------------------------
extern "C" void kernel_launch(void* const* d_in, const int* in_sizes, int n_in,
                              void* d_out, int out_size, void* d_ws, size_t ws_size,
                              hipStream_t stream) {
    const float* X = (const float*)d_in[0];
    const float* F = (const float*)d_in[1];
    const int*   N = (const int*)d_in[2];
    const float* Q = (const float*)d_in[3];
    const float* W = (const float*)d_in[4];
    float* Out = (float*)d_out;

    // workspace: packed bf16 W in WMMA-B lane layout (30*4*32*16*2 = 122880 B)
    __bf16* WB = (__bf16*)d_ws;
    pack_w_kernel<<<dim3((NKK * 4 * 32 * 16 + 255) / 256), dim3(256), 0, stream>>>(W, WB);

    dim3 grid(M_ / 16, B_);
    kpconv_kernel<<<grid, dim3(32), 0, stream>>>(X, F, N, Q, (const v8bf*)d_ws, Out);
}
